// RiboMIMO_18227841204374
// MI455X (gfx1250) — matrix operand order repaired and checked by reading, not compiled, not run on hardware
//
#include <hip/hip_runtime.h>
#include <math.h>

// ---------------------------------------------------------------------------
// Types for CDNA5 WMMA (wave32): 16x16x32 bf16 -> f32 accum
// ---------------------------------------------------------------------------
typedef __attribute__((ext_vector_type(16))) __bf16        v16bf;
typedef __attribute__((ext_vector_type(8)))  float         v8f;
typedef __attribute__((ext_vector_type(4)))  unsigned int  u32x4;

#define B_   64
#define T_   1024
#define H_   128
#define G3   384   // 3*H
#define NS   5
#define M_   (B_ * T_)   // 65536 rows

union AFrag {
    v16bf          v;
    unsigned short s[16];
    u32x4          q[2];
};

__device__ __forceinline__ unsigned short f2bf_bits(float f) {
    union { float f; unsigned u; } v; v.f = f;
    unsigned r = v.u + 0x7FFFu + ((v.u >> 16) & 1u);   // round-to-nearest-even
    return (unsigned short)(r >> 16);
}

__device__ __forceinline__ v8f wmma_bf16(const AFrag& a, const AFrag& b, v8f c) {
    return __builtin_amdgcn_wmma_f32_16x16x32_bf16(
        false, a.v, false, b.v, (short)0, c, false, false);
}

// ---------------------------------------------------------------------------
// Generic batched GEMM:  C[M,N] = A[M,K](bf16) * W[N,K](bf16)^T + bias (+relu)
// One 16x16 output tile per wave, K in chunks of 32 (one v_wmma each).
// Activations/weights pre-converted to bf16; L2 (192MB) holds A across N-tiles.
// ---------------------------------------------------------------------------
__global__ __launch_bounds__(256)
void wmma_gemm_bias(const unsigned short* __restrict__ Ab,
                    const unsigned short* __restrict__ Wb,
                    const float* __restrict__ bias,
                    float* __restrict__ C,
                    int M, int N, int K, int relu)
{
    const int lane = threadIdx.x & 31;
    const int wave = threadIdx.x >> 5;
    const int hlf  = lane >> 4;      // 0: lanes 0-15, 1: lanes 16-31
    const int lrow = lane & 15;
    const int ntiles = N >> 4;

    const int tile = blockIdx.x * (blockDim.x >> 5) + wave;
    if (tile >= (M >> 4) * ntiles) return;
    const int mBase = (tile / ntiles) * 16;
    const int nBase = (tile % ntiles) * 16;

    const unsigned short* arow = Ab + (size_t)(mBase + lrow) * K;
    const unsigned short* wrow = Wb + (size_t)(nBase + lrow) * K;

    v8f acc = {};
    for (int k0 = 0; k0 < K; k0 += 32) {
        AFrag a, b;
        // A 16x32 bf16 layout: elems 0-7 @ k0+hlf*8, elems 8-15 @ k0+16+hlf*8
        const unsigned short* pa = arow + k0 + hlf * 8;
        a.q[0] = *(const u32x4*)(pa);
        a.q[1] = *(const u32x4*)(pa + 16);
        // B 32x16 bf16 layout: lane=col, elems 0-7 @ k0+hlf*8, elems 8-15 @ k0+16+hlf*8
        const unsigned short* pb = wrow + k0 + hlf * 8;
        b.q[0] = *(const u32x4*)(pb);
        b.q[1] = *(const u32x4*)(pb + 16);
        acc = wmma_bf16(a, b, acc);
    }

    const int col = nBase + lrow;
    const float bv = bias[col];
#pragma unroll
    for (int r = 0; r < 8; ++r) {
        const int row = mBase + hlf * 8 + r;       // C: VGPR r -> M = hlf*8+r
        float v = acc[r] + bv;
        if (relu) v = fmaxf(v, 0.f);
        C[(size_t)row * N + col] = v;
    }
}

// ---------------------------------------------------------------------------
// Persistent bidirectional GRU scan. grid = {fwd, bwd}; one WGP each.
// h kept in LDS (f32 master + bf16 mirror). Per step: hg = h @ Whh^T + bhh
// via WMMA (32 wave-jobs, each = 3 gate tiles x 4 k-steps), then fused gate
// math in registers, masked state update, and output store. Whh (96KB bf16)
// streams from global and stays L2/WGP$ resident across all 1024 steps.
// ---------------------------------------------------------------------------
__global__ __launch_bounds__(512)
void gru_scan(const float* __restrict__ xg_f,
              const float* __restrict__ xg_b,
              const unsigned short* __restrict__ whh_f,
              const unsigned short* __restrict__ whh_b,
              const float* __restrict__ bhh_f,
              const float* __restrict__ bhh_b,
              const int* __restrict__ lengths,
              float* __restrict__ out)          // [B,T,2H], pre-zeroed
{
    __shared__ float          hS[B_ * H_];       // 32 KB f32 state
    __shared__ unsigned short h16[B_ * H_];      // 16 KB bf16 mirror
    __shared__ int            lenS[B_];

    const int dir = blockIdx.x;
    const float*          xg  = dir ? xg_b  : xg_f;
    const unsigned short* Whh = dir ? whh_b : whh_f;
    const float*          bhh = dir ? bhh_b : bhh_f;

    const int tid  = threadIdx.x;                // 512 = 16 waves
    const int lane = tid & 31;
    const int wave = tid >> 5;
    const int hlf  = lane >> 4;
    const int lrow = lane & 15;

    for (int i = tid; i < B_ * H_; i += 512) { hS[i] = 0.f; h16[i] = 0; }
    if (tid < B_) lenS[tid] = lengths[tid];
    __syncthreads();

    float hnew[16];

    for (int step = 0; step < T_; ++step) {
        // ---- phase 1: compute (reads h, no writes) ----
#pragma unroll
        for (int job = 0; job < 2; ++job) {
            const int jid = wave * 2 + job;        // 0..31
            const int mt  = jid >> 3;              // 4 row tiles (B=64)
            const int jt  = jid & 7;               // 8 col tiles (H=128)
            const int mBase = mt * 16;
            const int j     = jt * 16 + lrow;      // column within a gate

            v8f acc0 = {}, acc1 = {}, acc2 = {};
            const unsigned short* hrow = &h16[(mBase + lrow) * H_];
            const unsigned short* w0r  = Whh + (size_t)(j)        * H_;
            const unsigned short* w1r  = Whh + (size_t)(H_  + j)  * H_;
            const unsigned short* w2r  = Whh + (size_t)(2*H_ + j) * H_;
#pragma unroll
            for (int k0 = 0; k0 < H_; k0 += 32) {
                AFrag a, b0, b1, b2;
                const unsigned short* pa = hrow + k0 + hlf * 8;
                a.q[0] = *(const u32x4*)(pa);
                a.q[1] = *(const u32x4*)(pa + 16);
                const int ko = k0 + hlf * 8;
                b0.q[0] = *(const u32x4*)(w0r + ko); b0.q[1] = *(const u32x4*)(w0r + ko + 16);
                b1.q[0] = *(const u32x4*)(w1r + ko); b1.q[1] = *(const u32x4*)(w1r + ko + 16);
                b2.q[0] = *(const u32x4*)(w2r + ko); b2.q[1] = *(const u32x4*)(w2r + ko + 16);
                acc0 = wmma_bf16(a, b0, acc0);
                acc1 = wmma_bf16(a, b1, acc1);
                acc2 = wmma_bf16(a, b2, acc2);
            }
            // fused gate math (torch GRU: r,z,n ordering; frozen state at pad)
#pragma unroll
            for (int r = 0; r < 8; ++r) {
                const int b = mBase + hlf * 8 + r;
                const int L = lenS[b];
                const float hv = hS[b * H_ + j];
                float v = hv;
                if (step < L) {
                    const int t = dir ? (L - 1 - step) : step;
                    const float* xgp = xg + ((size_t)b * T_ + t) * G3;
                    const float xr = xgp[j], xz = xgp[H_ + j], xn = xgp[2*H_ + j];
                    const float hr = acc0[r] + bhh[j];
                    const float hz = acc1[r] + bhh[H_ + j];
                    const float hn = acc2[r] + bhh[2*H_ + j];
                    const float rg = 1.f / (1.f + __expf(-(xr + hr)));
                    const float zg = 1.f / (1.f + __expf(-(xz + hz)));
                    const float ng = tanhf(xn + rg * hn);
                    v = (1.f - zg) * ng + zg * hv;
                }
                hnew[job * 8 + r] = v;
            }
        }
        __syncthreads();
        // ---- phase 2: commit state + store output ----
#pragma unroll
        for (int job = 0; job < 2; ++job) {
            const int jid = wave * 2 + job;
            const int mt  = jid >> 3;
            const int jt  = jid & 7;
            const int j   = jt * 16 + lrow;
#pragma unroll
            for (int r = 0; r < 8; ++r) {
                const int b = mt * 16 + hlf * 8 + r;
                const int L = lenS[b];
                if (step < L) {
                    const float v = hnew[job * 8 + r];
                    hS [b * H_ + j] = v;
                    h16[b * H_ + j] = f2bf_bits(v);
                    const int t = dir ? (L - 1 - step) : step;
                    out[((size_t)b * T_ + t) * (2 * H_) + dir * H_ + j] = v;
                }
            }
        }
        __syncthreads();
    }
}

// ---------------------------------------------------------------------------
// Heads: y = f.reg_w + reg_b ; z = f.cla_w^T + cla_b   (K=128, thread/row)
// ---------------------------------------------------------------------------
__global__ __launch_bounds__(256)
void heads_kernel(const float* __restrict__ f,
                  const float* __restrict__ reg_w, const float* __restrict__ reg_b,
                  const float* __restrict__ cla_w, const float* __restrict__ cla_b,
                  float* __restrict__ y, float* __restrict__ z)
{
    const int row = blockIdx.x * blockDim.x + threadIdx.x;
    if (row >= M_) return;
    const float4* fr = (const float4*)(f + (size_t)row * H_);
    float accy = 0.f, accz[NS] = {0.f, 0.f, 0.f, 0.f, 0.f};
#pragma unroll 8
    for (int i = 0; i < H_ / 4; ++i) {
        const float4 fv = fr[i];
        const float4 rw = ((const float4*)reg_w)[i];
        accy += fv.x*rw.x + fv.y*rw.y + fv.z*rw.z + fv.w*rw.w;
#pragma unroll
        for (int s = 0; s < NS; ++s) {
            const float4 cw = ((const float4*)(cla_w + s * H_))[i];
            accz[s] += fv.x*cw.x + fv.y*cw.y + fv.z*cw.z + fv.w*cw.w;
        }
    }
    y[row] = accy + reg_b[0];
#pragma unroll
    for (int s = 0; s < NS; ++s) z[(size_t)row * NS + s] = accz[s] + cla_b[s];
}

// ---------------------------------------------------------------------------
// CRF NLL: 64 independent 5-state lattices, one lane each.
// ---------------------------------------------------------------------------
__global__ __launch_bounds__(64)
void crf_kernel(const float* __restrict__ z,        // [B,T,5]
                const int* __restrict__ tags,
                const int* __restrict__ lengths,
                const float* __restrict__ transition,
                float* __restrict__ loss_out)
{
    __shared__ float tr[NS * NS];
    __shared__ float vals[B_];
    const int tid = threadIdx.x;
    if (tid < NS * NS) tr[tid] = transition[tid];
    __syncthreads();

    if (tid < B_) {
        const int L = lengths[tid];
        float fv[NS];
#pragma unroll
        for (int s = 0; s < NS; ++s) fv[s] = (s == 0) ? 0.f : -1e6f;  // START=0
        float sc = 0.f; int prev = 0;
        const float* zb = z + (size_t)tid * T_ * NS;
        const int*   tg = tags + (size_t)tid * T_;
        for (int t = 0; t < T_; ++t) {
            if (t < L) {
                const float* e = zb + (size_t)t * NS;
                float nv[NS];
#pragma unroll
                for (int i = 0; i < NS; ++i) {
                    float m = -1e30f;
#pragma unroll
                    for (int jq = 0; jq < NS; ++jq) m = fmaxf(m, fv[jq] + tr[i*NS + jq]);
                    float ssum = 0.f;
#pragma unroll
                    for (int jq = 0; jq < NS; ++jq) ssum += __expf(fv[jq] + tr[i*NS + jq] - m);
                    nv[i] = e[i] + m + __logf(ssum);
                }
#pragma unroll
                for (int i = 0; i < NS; ++i) fv[i] = nv[i];
                const int tgt = tg[t];
                sc += tr[tgt * NS + prev] + e[tgt];   // transition[to, from]
                prev = tgt;
            }
        }
        sc += tr[(NS - 1) * NS + tg[L - 1]];          // STOP row
        float m = -1e30f;
#pragma unroll
        for (int s = 0; s < NS; ++s) m = fmaxf(m, fv[s] + tr[(NS-1)*NS + s]);
        float ssum = 0.f;
#pragma unroll
        for (int s = 0; s < NS; ++s) ssum += __expf(fv[s] + tr[(NS-1)*NS + s] - m);
        vals[tid] = sc - (m + __logf(ssum));
    }
    __syncthreads();
    if (tid == 0) {
        float a = 0.f;
        for (int i = 0; i < B_; ++i) a += vals[i];
        loss_out[0] = -a / (float)B_;
    }
}

// ---------------------------------------------------------------------------
// Small utilities
// ---------------------------------------------------------------------------
__global__ void f32_to_bf16(const float* __restrict__ s, unsigned short* __restrict__ d, int n) {
    const int i = blockIdx.x * blockDim.x + threadIdx.x;
    if (i < n) d[i] = f2bf_bits(s[i]);
}

__global__ void zero_f4(u32x4* p, size_t n4) {
    size_t i = (size_t)blockIdx.x * blockDim.x + threadIdx.x;
    const size_t st = (size_t)gridDim.x * blockDim.x;
    const u32x4 zz = {0u, 0u, 0u, 0u};
    for (; i < n4; i += st) p[i] = zz;
}

// ---------------------------------------------------------------------------
extern "C" void kernel_launch(void* const* d_in, const int* in_sizes, int n_in,
                              void* d_out, int out_size, void* d_ws, size_t ws_size,
                              hipStream_t stream) {
    (void)in_sizes; (void)n_in; (void)ws_size;

    const float* x = (const float*)d_in[0];
    // rnn params flattened: (layer, dir) x (Wih, Whh, bih, bhh)
    const float* Wih_p[4] = { (const float*)d_in[1],  (const float*)d_in[5],
                              (const float*)d_in[9],  (const float*)d_in[13] };
    const float* Whh_p[4] = { (const float*)d_in[2],  (const float*)d_in[6],
                              (const float*)d_in[10], (const float*)d_in[14] };
    const float* bih_p[4] = { (const float*)d_in[3],  (const float*)d_in[7],
                              (const float*)d_in[11], (const float*)d_in[15] };
    const float* bhh_p[4] = { (const float*)d_in[4],  (const float*)d_in[8],
                              (const float*)d_in[12], (const float*)d_in[16] };
    const float* fc_w  = (const float*)d_in[17];
    const float* fc_b  = (const float*)d_in[18];
    const float* reg_w = (const float*)d_in[19];
    const float* reg_b = (const float*)d_in[20];
    const float* cla_w = (const float*)d_in[21];
    const float* cla_b = (const float*)d_in[22];
    const float* trans = (const float*)d_in[23];
    const int* lengths = (const int*)d_in[24];
    const int* tags    = (const int*)d_in[25];

    // ---- workspace layout (bytes) ----
    char* ws = (char*)d_ws;
    float* hbuf0 = (float*)(ws + 0);                       //  64 MB  [B,T,256]
    float* hbuf1 = (float*)(ws + 67108864);                //  64 MB
    float* xg_f  = (float*)(ws + 134217728);               //  96 MB  [B,T,384]
    float* xg_b  = (float*)(ws + 234881024);               //  96 MB
    float* fbuf  = (float*)(ws + 335544320);               //  32 MB  [B,T,128]
    unsigned short* abuf = (unsigned short*)(ws + 369098752); // 32 MB bf16 acts
    unsigned short* wb   = (unsigned short*)(ws + 402653184); // bf16 weights
    unsigned short* wih0f = wb + 0;       // 384*64
    unsigned short* whh0f = wb + 24576;   // 384*128
    unsigned short* wih0b = wb + 73728;
    unsigned short* whh0b = wb + 98304;
    unsigned short* wih1f = wb + 147456;  // 384*256
    unsigned short* whh1f = wb + 245760;
    unsigned short* wih1b = wb + 294912;
    unsigned short* whh1b = wb + 393216;
    unsigned short* fcwb  = wb + 442368;  // 128*256

    auto conv = [&](const float* s, unsigned short* d, int n) {
        f32_to_bf16<<<(n + 255) / 256, 256, 0, stream>>>(s, d, n);
    };
    conv(Wih_p[0], wih0f, G3 * 64);  conv(Whh_p[0], whh0f, G3 * H_);
    conv(Wih_p[1], wih0b, G3 * 64);  conv(Whh_p[1], whh0b, G3 * H_);
    conv(Wih_p[2], wih1f, G3 * 256); conv(Whh_p[2], whh1f, G3 * H_);
    conv(Wih_p[3], wih1b, G3 * 256); conv(Whh_p[3], whh1b, G3 * H_);
    conv(fc_w, fcwb, H_ * 256);

    // zero hbuf0+hbuf1 (contiguous): padded-tail positions must be 0
    zero_f4<<<2048, 256, 0, stream>>>((u32x4*)hbuf0, (size_t)(2 * M_ * 256) / 4);

    float* y  = (float*)d_out;
    float* zc = y + M_;
    float* lp = y + (out_size - 1);

    // ---- layer 0 (K=64) ----
    conv(x, abuf, M_ * 64);
    wmma_gemm_bias<<<12288, 256, 0, stream>>>(abuf, wih0f, bih_p[0], xg_f, M_, G3, 64, 0);
    wmma_gemm_bias<<<12288, 256, 0, stream>>>(abuf, wih0b, bih_p[1], xg_b, M_, G3, 64, 0);
    gru_scan<<<2, 512, 0, stream>>>(xg_f, xg_b, whh0f, whh0b, bhh_p[0], bhh_p[1], lengths, hbuf0);

    // ---- layer 1 (K=256) ----
    conv(hbuf0, abuf, M_ * 256);
    wmma_gemm_bias<<<12288, 256, 0, stream>>>(abuf, wih1f, bih_p[2], xg_f, M_, G3, 256, 0);
    wmma_gemm_bias<<<12288, 256, 0, stream>>>(abuf, wih1b, bih_p[3], xg_b, M_, G3, 256, 0);
    gru_scan<<<2, 512, 0, stream>>>(xg_f, xg_b, whh1f, whh1b, bhh_p[2], bhh_p[3], lengths, hbuf1);

    // ---- FC + ReLU (K=256, N=128) ----
    conv(hbuf1, abuf, M_ * 256);
    wmma_gemm_bias<<<4096, 256, 0, stream>>>(abuf, fcwb, fc_b, fbuf, M_, H_, 256, 1);

    heads_kernel<<<(M_ + 255) / 256, 256, 0, stream>>>(fbuf, reg_w, reg_b, cla_w, cla_b, y, zc);
    crf_kernel<<<1, 64, 0, stream>>>(zc, tags, lengths, trans, lp);
}
